// CombNetHE_8117488190156
// MI455X (gfx1250) — matrix operand order repaired and checked by reading, not compiled, not run on hardware
//
#include <hip/hip_runtime.h>
#include <hip/hip_bf16.h>
#include <math.h>

typedef __bf16 bf16_t;
typedef __attribute__((ext_vector_type(16))) __bf16 v16bf;
typedef __attribute__((ext_vector_type(8)))  __bf16 v8bf;
typedef __attribute__((ext_vector_type(8)))  float  v8f;

#define B_SZ   16384
#define D_IN   1024
#define D_H    4096
#define NCLS   10
#define TAU_F  0.5f
#define T1_IT  3
#define T2_IT  3

#define BM   128
#define BN   128
#define KT   32
#define LDA  40      /* KT + 8 bf16 pad -> 80B row stride, avoids LDS bank conflicts */
#define HPAD 136     /* 128 + 8 pad for h tile staging */
#define NKT  (D_IN / KT)

/* gfx1250 async global->LDS copy (VGLOBAL encoding: vdst = LDS byte offset
 * within the workgroup allocation, vaddr = 64-bit global address).
 * Tracked by ASYNCcnt; drained with s_wait_asynccnt. */
#define ASYNC_LD_B128(ldsOff, gptr)                                   \
    asm volatile("global_load_async_to_lds_b128 %0, %1, off"          \
                 :: "v"(ldsOff), "v"(gptr) : "memory")
#define WAIT_ASYNC0() asm volatile("s_wait_asynccnt 0x0" ::: "memory")

union AFrag { v16bf v; v8bf h[2]; };

// ---------------------------------------------------------------- utilities

__global__ void zero_f32(float* __restrict__ p, int n) {
    int i = blockIdx.x * blockDim.x + threadIdx.x;
    if (i < n) p[i] = 0.0f;
}

__global__ void f32_to_bf16_4(const float* __restrict__ src,
                              bf16_t* __restrict__ dst, int n4) {
    int i = blockIdx.x * blockDim.x + threadIdx.x;
    if (i < n4) {
        float4 v = ((const float4*)src)[i];
        union { bf16_t b[4]; uint2 u; } t;
        t.b[0] = (bf16_t)v.x; t.b[1] = (bf16_t)v.y;
        t.b[2] = (bf16_t)v.z; t.b[3] = (bf16_t)v.w;
        ((uint2*)dst)[i] = t.u;
    }
}

// W1 is [D_IN][D_H] row-major f32; produce W1^T [D_H][D_IN] bf16 so that a
// WMMA B-fragment (col n, contiguous K) is two contiguous 8-element K runs.
__global__ void transpose_f32_bf16(const float* __restrict__ W,
                                   bf16_t* __restrict__ WT) {
    __shared__ float tile[32][33];
    const int n0 = blockIdx.x * 32;   // D_H index
    const int k0 = blockIdx.y * 32;   // D_IN index
    const int tx = threadIdx.x, ty = threadIdx.y;   // 32 x 8
#pragma unroll
    for (int i = 0; i < 32; i += 8)
        tile[ty + i][tx] = W[(size_t)(k0 + ty + i) * D_H + n0 + tx];
    __syncthreads();
#pragma unroll
    for (int i = 0; i < 32; i += 8)
        WT[(size_t)(n0 + ty + i) * D_IN + k0 + tx] = (bf16_t)tile[tx][ty + i];
}

// ------------------------------------------------- fused GEMM1 + partial GEMM2
//
// grid = (B/128, D_H/128, 2 nets), block = 128 threads = 4 wave32.
// Wave grid 2x2: each wave owns a 64x64 piece = 4x4 WMMA 16x16 accum tiles,
// so each B-fragment reload feeds 4 WMMAs (amortizes LDS latency) and LDS
// traffic per FLOP is halved vs a 32x64 wave tile.
__global__ __launch_bounds__(128)
void gemm1_fused(const bf16_t* __restrict__ xb,
                 const bf16_t* __restrict__ w1ot, const bf16_t* __restrict__ w1ft,
                 const float*  __restrict__ b1o,  const float*  __restrict__ b1f,
                 const float*  __restrict__ w2o,  const float*  __restrict__ w2f,
                 float* __restrict__ lo, float* __restrict__ lf)
{
    // Single shared allocation => LDS offset of smem within the workgroup
    // allocation is 0, so plain byte offsets are valid async-copy LDS targets.
    __shared__ __align__(16) bf16_t smem[20480];  // 40 KB, reused for h staging

    const int net = blockIdx.z;
    const bf16_t* __restrict__ w1t    = net ? w1ft : w1ot;
    const float*  __restrict__ b1     = net ? b1f  : b1o;
    const float*  __restrict__ w2     = net ? w2f  : w2o;
    float*        __restrict__ logits = net ? lf   : lo;

    const int row_base = blockIdx.x * BM;
    const int col_base = blockIdx.y * BN;
    const int tid  = threadIdx.x;
    const int lane = tid & 31;
    const int wid  = tid >> 5;
    const int wm   = (wid & 1) * 64;   // wave row offset in tile
    const int wn   = (wid >> 1) * 64;  // wave col offset in tile

    // ---- staging: each thread async-copies four 16B chunks of A and of B/stage
    const int r0 = tid >> 2;              // rows r0, r0+32, r0+64, r0+96
    const int kc = (tid & 3) * 8;
    const bf16_t* gA[4];
    const bf16_t* gB[4];
    unsigned ldsA[4], ldsB[4];
#pragma unroll
    for (int q = 0; q < 4; ++q) {
        const int row = r0 + q * 32;
        gA[q] = xb  + (size_t)(row_base + row) * D_IN + kc;
        gB[q] = w1t + (size_t)(col_base + row) * D_IN + kc;
        ldsA[q] = (unsigned)((row * LDA + kc) * 2);           // sA at LDS byte 0
        ldsB[q] = 20480u + (unsigned)((row * LDA + kc) * 2);  // sB at LDS byte 20480
    }

    // preload stage 0 into buffer 0 (async global->LDS, ASYNCcnt tracked)
#pragma unroll
    for (int q = 0; q < 4; ++q) {
        ASYNC_LD_B128(ldsA[q], gA[q]);
        ASYNC_LD_B128(ldsB[q], gB[q]);
    }
    WAIT_ASYNC0();
    __syncthreads();

    v8f acc[4][4];
    const v8f vzero = {0.f, 0.f, 0.f, 0.f, 0.f, 0.f, 0.f, 0.f};
#pragma unroll
    for (int i = 0; i < 4; ++i)
#pragma unroll
        for (int j = 0; j < 4; ++j) acc[i][j] = vzero;

    // fragment addressing (ISA 7.12.2, 16-bit A 16x32 / B 32x16 wave32 layouts)
    const int fm     = lane & 15;
    const int khalfA = (lane >> 4) * 8;   // A: K split 0..7/16..23 vs 8..15/24..31
    const int kqB    = (lane >> 4) * 8;   // B: K split 0..7/16..23 vs 8..15/24..31

#pragma unroll 1
    for (int kt = 0; kt < NKT; ++kt) {
        const int cur = kt & 1;
        const int nxt = cur ^ 1;
        const unsigned nb = (unsigned)(nxt * 10240);   // stage byte offset

        if (kt + 1 < NKT) {   // async-copy next stage straight into LDS
            const size_t go = (size_t)(kt + 1) * KT;
#pragma unroll
            for (int q = 0; q < 4; ++q) {
                ASYNC_LD_B128(ldsA[q] + nb, gA[q] + go);
                ASYNC_LD_B128(ldsB[q] + nb, gB[q] + go);
            }
        }
        if (kt + 2 < NKT) {  // gfx1250 global_prefetch of the stage after next
            __builtin_prefetch(gA[0] + (size_t)(kt + 2) * KT, 0, 1);
            __builtin_prefetch(gB[0] + (size_t)(kt + 2) * KT, 0, 1);
        }

        const bf16_t* aBase = smem + cur * 5120;           // elements
        const bf16_t* bBase = smem + 10240 + cur * 5120;

        // A fragments loaded once per stage, each reused by 4 WMMAs
        AFrag afr[4];
#pragma unroll
        for (int im = 0; im < 4; ++im) {
            const bf16_t* p = aBase + (wm + im * 16 + fm) * LDA + khalfA;
            afr[im].h[0] = *(const v8bf*)(p);        // K = khalf .. +7
            afr[im].h[1] = *(const v8bf*)(p + 16);   // K = khalf+16 .. +23
        }
#pragma unroll
        for (int in = 0; in < 4; ++in) {
            AFrag bfr;
            const bf16_t* p = bBase + (wn + in * 16 + fm) * LDA + kqB;
            bfr.h[0] = *(const v8bf*)(p);            // K = kq .. +7
            bfr.h[1] = *(const v8bf*)(p + 16);       // K = kq+16 .. +23
#pragma unroll
            for (int im = 0; im < 4; ++im)
                acc[im][in] = __builtin_amdgcn_wmma_f32_16x16x32_bf16(
                    false, afr[im].v, false, bfr.v,
                    (short)0, acc[im][in], false, false);
        }

        if (kt + 1 < NKT) WAIT_ASYNC0();   // our async writes landed in LDS
        __syncthreads();                    // publish to all waves
    }

    // ---- epilogue: bias + ReLU, stage h tile (bf16) into reused LDS
    {
        const int rsel = lane >> 4;  // 0 => M=r, 1 => M=r+8 (C/D layout)
#pragma unroll
        for (int im = 0; im < 4; ++im) {
#pragma unroll
            for (int in = 0; in < 4; ++in) {
                const int ccol  = wn + in * 16 + fm;
                const float bias = b1[col_base + ccol];
#pragma unroll
                for (int r = 0; r < 8; ++r) {
                    const int mrow = wm + im * 16 + r + 8 * rsel;
                    float v = acc[im][in][r] + bias;
                    v = v > 0.0f ? v : 0.0f;
                    smem[mrow * HPAD + ccol] = (bf16_t)v;
                }
            }
        }
    }
    __syncthreads();

    // ---- fused layer-2 partial: this block's 128 cols -> 10-class partial logits
    const int m = tid;   // one thread per row, all 128 cols
    float a10[NCLS];
#pragma unroll
    for (int c = 0; c < NCLS; ++c) a10[c] = 0.0f;
    for (int j = 0; j < 128; ++j) {
        const float hv = (float)smem[m * HPAD + j];
        const float* w2p = w2 + (size_t)(col_base + j) * NCLS;
#pragma unroll
        for (int c = 0; c < NCLS; ++c) a10[c] = fmaf(hv, w2p[c], a10[c]);
    }
    float* lp = logits + (size_t)(row_base + m) * NCLS;
#pragma unroll
    for (int c = 0; c < NCLS; ++c) unsafeAtomicAdd(lp + c, a10[c]);
}

// --------------------------------------------------------------- finalize
__global__ void finalize_kernel(const float* __restrict__ lo,
                                const float* __restrict__ lf,
                                const float* __restrict__ b2o,
                                const float* __restrict__ b2f,
                                float* __restrict__ out)
{
    int r = blockIdx.x * blockDim.x + threadIdx.x;
    if (r >= B_SZ) return;

    float x1[NCLS], x2[NCLS];
    float m1 = -1e30f, m2 = -1e30f;
#pragma unroll
    for (int c = 0; c < NCLS; ++c) {
        x1[c] = lo[(size_t)r * NCLS + c] + b2o[c];
        x2[c] = lf[(size_t)r * NCLS + c] + b2f[c];
        m1 = fmaxf(m1, x1[c]);
        m2 = fmaxf(m2, x2[c]);
    }
    float s1 = 0.0f, s2 = 0.0f;
#pragma unroll
    for (int c = 0; c < NCLS; ++c) {
        x1[c] = __expf(x1[c] - m1); s1 += x1[c];
        x2[c] = __expf(x2[c] - m2); s2 += x2[c];
    }
    const float i1 = 1.0f / s1, i2 = 1.0f / s2;
#pragma unroll
    for (int c = 0; c < NCLS; ++c) { x1[c] *= i1; x2[c] *= i2; }

    // comp_max_tau(x1, tau=0.5, t1=3, t2=3)
    float res[NCLS + 1];
#pragma unroll
    for (int c = 0; c < NCLS; ++c) res[c] = x1[c];
    res[NCLS] = TAU_F;
#pragma unroll
    for (int i = 0; i < T1_IT; ++i) {
        float s = 0.0f;
#pragma unroll
        for (int c = 0; c <= NCLS; ++c) { res[c] *= res[c]; s += res[c]; }
        const float mm = (i == 0) ? (2.0f + TAU_F * TAU_F) : 2.0f;
        const float k  = 2.0f / mm;
        float a = 2.0f - k * s;
        float b = 1.0f - k * s;
#pragma unroll
        for (int t = 0; t < T2_IT; ++t) { b = b * b; a = a * (1.0f + b); }
        const float inv = k * a;
#pragma unroll
        for (int c = 0; c <= NCLS; ++c) res[c] *= inv;
    }
    const float cond = res[NCLS];
#pragma unroll
    for (int c = 0; c < NCLS; ++c)
        out[(size_t)r * NCLS + c] = x1[c] * (1.0f - cond) + x2[c] * cond;
}

// --------------------------------------------------------------- launch
extern "C" void kernel_launch(void* const* d_in, const int* in_sizes, int n_in,
                              void* d_out, int out_size, void* d_ws, size_t ws_size,
                              hipStream_t stream) {
    const float* x   = (const float*)d_in[0];
    const float* W1o = (const float*)d_in[1];
    const float* b1o = (const float*)d_in[2];
    const float* W2o = (const float*)d_in[3];
    const float* b2o = (const float*)d_in[4];
    const float* W1f = (const float*)d_in[5];
    const float* b1f = (const float*)d_in[6];
    const float* W2f = (const float*)d_in[7];
    const float* b2f = (const float*)d_in[8];
    float* out = (float*)d_out;

    char* ws = (char*)d_ws;
    bf16_t* xb   = (bf16_t*)(ws);                         // 32 MB
    bf16_t* w1ot = (bf16_t*)(ws + 33554432ull);           // 8 MB
    bf16_t* w1ft = (bf16_t*)(ws + 41943040ull);           // 8 MB
    float*  lo   = (float*) (ws + 50331648ull);           // 640 KB
    float*  lf   = (float*) (ws + 50331648ull + 655360ull);

    const int nlog = 2 * B_SZ * NCLS;      // lo and lf are contiguous
    zero_f32<<<dim3((nlog + 255) / 256), dim3(256), 0, stream>>>(lo, nlog);

    f32_to_bf16_4<<<dim3((B_SZ * D_IN / 4) / 256), dim3(256), 0, stream>>>(
        x, xb, B_SZ * D_IN / 4);

    transpose_f32_bf16<<<dim3(D_H / 32, D_IN / 32), dim3(32, 8), 0, stream>>>(W1o, w1ot);
    transpose_f32_bf16<<<dim3(D_H / 32, D_IN / 32), dim3(32, 8), 0, stream>>>(W1f, w1ft);

    gemm1_fused<<<dim3(B_SZ / BM, D_H / BN, 2), dim3(128), 0, stream>>>(
        xb, w1ot, w1ft, b1o, b1f, W2o, W2f, lo, lf);

    finalize_kernel<<<dim3(B_SZ / 256), dim3(256), 0, stream>>>(lo, lf, b2o, b2f, out);
}
